// BoundaryConvLayer_88983132439348
// MI455X (gfx1250) — compile-verified, matrix operand order checked by reading, not run on hardware
//
#include <hip/hip_runtime.h>
#include <stdint.h>

// gfx1250 async global->LDS DMA path (ASYNCcnt-tracked) — assembled OK in R2.
#define USE_ASYNC_LDS 1

// ---------------------------------------------------------------------------
// Types for WMMA bf16 (CDNA5 / gfx1250, wave32)
// ---------------------------------------------------------------------------
typedef __attribute__((ext_vector_type(16))) __bf16 v16bf;
typedef __attribute__((ext_vector_type(8)))  __bf16 v8bf;
typedef __attribute__((ext_vector_type(8)))  float  v8f;

__device__ __forceinline__ unsigned short f2bf(float f) {
  unsigned int u = __float_as_uint(f);
  unsigned int r = u + 0x7FFFu + ((u >> 16) & 1u);
  return (unsigned short)(r >> 16);
}

enum { ACT_NONE = 0, ACT_SOFTPLUS = 1, ACT_GELU = 2 };

__device__ __forceinline__ float apply_act(float v, int ACT) {
  if (ACT == ACT_SOFTPLUS) {
    return (v > 20.f) ? v : log1pf(expf(v));
  } else if (ACT == ACT_GELU) {
    return 0.5f * v * (1.f + erff(v * 0.70710678118654752f));
  }
  return v;
}

// ---------------------------------------------------------------------------
// Weight prep: f32 [K x Ncols] row-major  ->  bf16 [Ncols x K] (transposed)
// ---------------------------------------------------------------------------
__global__ void convert_transpose_w(const float* __restrict__ src,
                                    unsigned short* __restrict__ dst,
                                    int K, int Ncols) {
  int idx = blockIdx.x * blockDim.x + threadIdx.x;
  if (idx >= K * Ncols) return;
  int k = idx / Ncols;
  int c = idx - k * Ncols;
  dst[(size_t)c * K + k] = f2bf(src[idx]);
}

// x -> bf16 copy, and zero the aggregation buffer (per-call, graph-safe)
__global__ void init_x_agg(const float* __restrict__ x,
                           unsigned short* __restrict__ xb,
                           float* __restrict__ agg, int total) {
  int idx = blockIdx.x * blockDim.x + threadIdx.x;
  if (idx >= total) return;
  xb[idx] = f2bf(x[idx]);
  agg[idx] = 0.f;
}

// ---------------------------------------------------------------------------
// Edge scatter: agg[dst] += x[src]   (E edges, D=128, 4 floats per thread)
// ---------------------------------------------------------------------------
__global__ void scatter_add(const float* __restrict__ x,
                            const long long* __restrict__ ei,
                            float* __restrict__ agg, long long E) {
  long long idx = (long long)blockIdx.x * blockDim.x + threadIdx.x;
  if (idx >= E * 32ll) return;
  long long e = idx >> 5;
  int c = (int)(idx & 31) << 2;
  long long s = ei[e];
  long long d = ei[E + e];
  const float4 v = *(const float4*)(x + s * 128 + c);
  float* a = agg + d * 128 + c;
  atomicAdd(a + 0, v.x);
  atomicAdd(a + 1, v.y);
  atomicAdd(a + 2, v.z);
  atomicAdd(a + 3, v.w);
}

// ---------------------------------------------------------------------------
// WMMA GEMM:  C[n,Ncols] = act(A[n,K] @ W[K,Ncols] + bias)
//   Block = 8 waves = 256 rows x 64 cols of C.
//   B tile [64 x K] bf16 staged once per block into LDS via async DMA.
//   Each wave computes 2 row tiles x 4 col tiles (8 accumulators):
//     - per K-step: 4 global b128 (A), 8 ds b128 (B), 8 WMMAs
//     - B fragment reused across both row tiles -> 1 ds_load_b128 per WMMA
//     - all B loads issued before the WMMA burst (staggered dscnt waits)
// Fragment layouts per CDNA5 ISA 7.12.2.
// ---------------------------------------------------------------------------
template <int ACT>
__global__ __launch_bounds__(256) void gemm_wmma(
    const unsigned short* __restrict__ Abits,
    const unsigned short* __restrict__ Wtbits,
    const float* __restrict__ bias,
    float* __restrict__ Cf,            // optional f32 output
    unsigned short* __restrict__ Cb,   // optional bf16 output
    int n, int K, int Ncols) {
  extern __shared__ unsigned short bsh[];  // [64][K+8] bf16 (padded rows)
  const int lane = threadIdx.x & 31;
  const int wave = threadIdx.x >> 5;
  const int r0 = blockIdx.x * 256;
  const int c0 = blockIdx.y * 64;
  const int KP = K + 8;                    // 16B row pad keeps banks spread

  // ---- stage B tile [64 x K] into LDS, one 16B chunk per thread-iter ----
  const __bf16* Wt = reinterpret_cast<const __bf16*>(Wtbits);
  const int chunks = (64 * K) >> 3;        // number of 8-element chunks
  for (int i = threadIdx.x; i < chunks; i += 256) {
    int el = i << 3;
    int c = el / K;
    int k = el - c * K;
    const __bf16* g = Wt + (size_t)(c0 + c) * K + k;
#if defined(USE_ASYNC_LDS)
    unsigned lds_addr = (unsigned)(size_t)&bsh[c * KP + k];
    asm volatile("global_load_async_to_lds_b128 %0, %1, off"
                 :: "v"(lds_addr), "v"(g)
                 : "memory");
#else
    *(v8bf*)&bsh[c * KP + k] = *(const v8bf*)g;
#endif
  }
#if defined(USE_ASYNC_LDS)
  asm volatile("s_wait_asynccnt 0x0" ::: "memory");
#endif
  __syncthreads();

  const int half = lane >> 4;              // 0: lanes 0-15, 1: lanes 16-31
  const int lid  = lane & 15;
  const int koff = half ? 8 : 0;

  int arow0 = r0 + wave * 32 + lid;        // row tile 0
  int arow1 = arow0 + 16;                  // row tile 1
  if (arow0 >= n) arow0 = n - 1;           // safe load clamps (stores guarded)
  if (arow1 >= n) arow1 = n - 1;

  const __bf16* A  = reinterpret_cast<const __bf16*>(Abits);
  const __bf16* Bs = reinterpret_cast<const __bf16*>(bsh);
  const size_t abase0 = (size_t)arow0 * K;
  const size_t abase1 = (size_t)arow1 * K;

  v8f acc[2][4] = {};
  for (int k0 = 0; k0 < K; k0 += 32) {
    // A fragments for both row tiles
    v8bf a0lo = *(const v8bf*)(A + abase0 + k0 + koff);
    v8bf a0hi = *(const v8bf*)(A + abase0 + k0 + 16 + koff);
    v8bf a1lo = *(const v8bf*)(A + abase1 + k0 + koff);
    v8bf a1hi = *(const v8bf*)(A + abase1 + k0 + 16 + koff);
    v16bf av0, av1;
#pragma unroll
    for (int i = 0; i < 8; ++i) {
      av0[i] = a0lo[i]; av0[i + 8] = a0hi[i];
      av1[i] = a1lo[i]; av1[i + 8] = a1hi[i];
    }
    // Issue all B loads first so LDS latency overlaps the WMMA burst
    v16bf bv[4];
#pragma unroll
    for (int ct = 0; ct < 4; ++ct) {
      const __bf16* bp = Bs + (size_t)(ct * 16 + lid) * KP + k0 + koff;
      v8bf blo = *(const v8bf*)bp;         // ds_load_b128
      v8bf bhi = *(const v8bf*)(bp + 16);  // ds_load_b128
#pragma unroll
      for (int i = 0; i < 8; ++i) { bv[ct][i] = blo[i]; bv[ct][i + 8] = bhi[i]; }
    }
#pragma unroll
    for (int ct = 0; ct < 4; ++ct) {
      acc[0][ct] = __builtin_amdgcn_wmma_f32_16x16x32_bf16(
          false, av0, false, bv[ct], (short)0, acc[0][ct], false, false);
      acc[1][ct] = __builtin_amdgcn_wmma_f32_16x16x32_bf16(
          false, av1, false, bv[ct], (short)0, acc[1][ct], false, false);
    }
  }

#pragma unroll
  for (int rt = 0; rt < 2; ++rt) {
#pragma unroll
    for (int ct = 0; ct < 4; ++ct) {
      const int gcol = c0 + ct * 16 + lid;
      const float bs = bias[gcol];
#pragma unroll
      for (int r = 0; r < 8; ++r) {
        const int grow = r0 + wave * 32 + rt * 16 + (half ? r + 8 : r);
        if (grow >= n) continue;
        float v = apply_act(acc[rt][ct][r] + bs, ACT);
        const size_t o = (size_t)grow * Ncols + gcol;
        if (Cf) Cf[o] = v;
        if (Cb) Cb[o] = f2bf(v);
      }
    }
  }
}

// ---------------------------------------------------------------------------
// Row-wise fuse: gamma = LN(gpre; g_rb,b_rb);
//                h = (rate*agg + gamma) / (1 + rate*degree + 1e-4) -> bf16
// ---------------------------------------------------------------------------
__global__ void fuse_h(const float* __restrict__ gpre,
                       const float* __restrict__ rate,
                       const float* __restrict__ agg,
                       const float* __restrict__ degree,
                       const float* __restrict__ g_rb,
                       const float* __restrict__ b_rb,
                       unsigned short* __restrict__ hb, int n) {
  const int row = blockIdx.x;
  const int tid = threadIdx.x;
  if (row >= n) return;
  __shared__ float red[128];
  const size_t o = (size_t)row * 128 + tid;
  const float v = gpre[o];
  red[tid] = v;
  __syncthreads();
  for (int s = 64; s > 0; s >>= 1) {
    if (tid < s) red[tid] += red[tid + s];
    __syncthreads();
  }
  const float mean = red[0] * (1.f / 128.f);
  __syncthreads();
  const float dv = v - mean;
  red[tid] = dv * dv;
  __syncthreads();
  for (int s = 64; s > 0; s >>= 1) {
    if (tid < s) red[tid] += red[tid + s];
    __syncthreads();
  }
  const float var = red[0] * (1.f / 128.f);
  const float gamma = dv * rsqrtf(var + 1e-5f) * g_rb[tid] + b_rb[tid];
  const float r = rate[o];
  const float a = agg[o];
  const float dg = degree[row];
  const float h = (r * a + gamma) / (1.f + r * dg + 1e-4f);
  hb[o] = f2bf(h);
}

// out += LayerNorm(x; g_n, b_n)   (residual path)
__global__ void ln_residual_add(const float* __restrict__ x,
                                const float* __restrict__ g_n,
                                const float* __restrict__ b_n,
                                float* __restrict__ out, int n) {
  const int row = blockIdx.x;
  const int tid = threadIdx.x;
  if (row >= n) return;
  __shared__ float red[128];
  const size_t o = (size_t)row * 128 + tid;
  const float v = x[o];
  red[tid] = v;
  __syncthreads();
  for (int s = 64; s > 0; s >>= 1) {
    if (tid < s) red[tid] += red[tid + s];
    __syncthreads();
  }
  const float mean = red[0] * (1.f / 128.f);
  __syncthreads();
  const float dv = v - mean;
  red[tid] = dv * dv;
  __syncthreads();
  for (int s = 64; s > 0; s >>= 1) {
    if (tid < s) red[tid] += red[tid + s];
    __syncthreads();
  }
  const float var = red[0] * (1.f / 128.f);
  out[o] += dv * rsqrtf(var + 1e-5f) * g_n[tid] + b_n[tid];
}

// ---------------------------------------------------------------------------
// Host-side orchestration
// ---------------------------------------------------------------------------
extern "C" void kernel_launch(void* const* d_in, const int* in_sizes, int n_in,
                              void* d_out, int out_size, void* d_ws, size_t ws_size,
                              hipStream_t stream) {
  const float*     x   = (const float*)d_in[0];
  const long long* ei  = (const long long*)d_in[1];
  const float*     deg = (const float*)d_in[2];
  const float*     Wr  = (const float*)d_in[3];
  const float*     br  = (const float*)d_in[4];
  const float*     Wb1 = (const float*)d_in[5];
  const float*     bb1 = (const float*)d_in[6];
  const float*     Wb2 = (const float*)d_in[7];
  const float*     bb2 = (const float*)d_in[8];
  const float*     g_rb= (const float*)d_in[9];
  const float*     b_rb= (const float*)d_in[10];
  const float*     W1  = (const float*)d_in[11];
  const float*     b1  = (const float*)d_in[12];
  const float*     W2  = (const float*)d_in[13];
  const float*     b2  = (const float*)d_in[14];
  const float*     g_n = (const float*)d_in[15];
  const float*     b_n = (const float*)d_in[16];
  float* out = (float*)d_out;

  const int D = 128;
  const int n = in_sizes[0] / D;            // 100000
  const int H = in_sizes[5] / D;            // 512
  const long long E = in_sizes[1] / 2;      // 1.6e6

  // Workspace carve-up (256B aligned)
  char* w = (char*)d_ws;
  auto carve = [&](size_t bytes) -> void* {
    void* p = (void*)w;
    w += (bytes + 255) & ~(size_t)255;
    return p;
  };
  float*          agg   = (float*)carve((size_t)n * D * 4);
  unsigned short* xb    = (unsigned short*)carve((size_t)n * D * 2); // reused as h bf16
  float*          rate  = (float*)carve((size_t)n * D * 4);
  float*          gpre  = (float*)carve((size_t)n * D * 4);
  unsigned short* h1b   = (unsigned short*)carve((size_t)n * H * 2); // reused for both H-wide acts
  unsigned short* wr_t  = (unsigned short*)carve((size_t)D * D * 2);
  unsigned short* wb1_t = (unsigned short*)carve((size_t)D * H * 2);
  unsigned short* wb2_t = (unsigned short*)carve((size_t)H * D * 2);
  unsigned short* w1_t  = (unsigned short*)carve((size_t)D * H * 2);
  unsigned short* w2_t  = (unsigned short*)carve((size_t)H * D * 2);

  const int TB = 256;
  auto cdiv = [](long long a, long long b) -> unsigned int {
    return (unsigned int)((a + b - 1) / b);
  };

  // 1. Weight convert + transpose (bf16, [Ncols x K])
  convert_transpose_w<<<cdiv((long long)D * D, TB), TB, 0, stream>>>(Wr,  wr_t,  D, D);
  convert_transpose_w<<<cdiv((long long)D * H, TB), TB, 0, stream>>>(Wb1, wb1_t, D, H);
  convert_transpose_w<<<cdiv((long long)H * D, TB), TB, 0, stream>>>(Wb2, wb2_t, H, D);
  convert_transpose_w<<<cdiv((long long)D * H, TB), TB, 0, stream>>>(W1,  w1_t,  D, H);
  convert_transpose_w<<<cdiv((long long)H * D, TB), TB, 0, stream>>>(W2,  w2_t,  H, D);

  // 2. x -> bf16, agg = 0
  init_x_agg<<<cdiv((long long)n * D, TB), TB, 0, stream>>>(x, xb, agg, n * D);

  // 3. Edge scatter (atomic f32)
  scatter_add<<<cdiv(E * 32, TB), TB, 0, stream>>>(x, ei, agg, E);

  const unsigned int rb = cdiv(n, 256);               // row blocks (256 rows each)
  auto smem = [](int K) { return (size_t)64 * (K + 8) * 2; };

  // 4. rate = softplus(x @ Wr + br)           [n x D], f32
  gemm_wmma<ACT_SOFTPLUS><<<dim3(rb, D / 64), TB, smem(D), stream>>>(
      xb, wr_t, br, rate, nullptr, n, D, D);
  // 5. t1 = softplus(x @ Wb1 + bb1)           [n x H], bf16
  gemm_wmma<ACT_SOFTPLUS><<<dim3(rb, H / 64), TB, smem(D), stream>>>(
      xb, wb1_t, bb1, nullptr, h1b, n, D, H);
  // 6. gpre = t1 @ Wb2 + bb2                  [n x D], f32
  gemm_wmma<ACT_NONE><<<dim3(rb, D / 64), TB, smem(H), stream>>>(
      h1b, wb2_t, bb2, gpre, nullptr, n, H, D);
  // 7. h = (rate*agg + LN(gpre)) / (1 + rate*deg + eps) -> bf16 (reuse xb)
  fuse_h<<<(unsigned int)n, 128, 0, stream>>>(gpre, rate, agg, deg, g_rb, b_rb, xb, n);
  // 8. g = gelu(h @ W1 + b1)                  [n x H], bf16 (reuse h1b)
  gemm_wmma<ACT_GELU><<<dim3(rb, H / 64), TB, smem(D), stream>>>(
      xb, w1_t, b1, nullptr, h1b, n, D, H);
  // 9. out = g @ W2 + b2                      [n x D], f32 -> d_out
  gemm_wmma<ACT_NONE><<<dim3(rb, D / 64), TB, smem(H), stream>>>(
      h1b, w2_t, b2, out, nullptr, n, H, D);
  // 10. out += LN(x; g_n, b_n)
  ln_residual_add<<<(unsigned int)n, 128, 0, stream>>>(x, g_n, b_n, out, n);
}
